// MultiheadAttention_40862318854307
// MI455X (gfx1250) — matrix operand order repaired and checked by reading, not compiled, not run on hardware
//
#include <hip/hip_runtime.h>
#include <hip/hip_bf16.h>

// MultiheadAttention forward for MI455X (gfx1250, wave32, WMMA).
//  B=2, T1=T2=2048, D=1024, H=16, HD=64
//  d_out = out [B,T1,D] f32  ++  attn [B,H,T1,T2] f32
//
// Pipeline (f16 operands / f32 accum, v_wmma_f32_16x16x32_f16):
//   1) q/k/v projections -> f16 ws (q folded with 1/sqrt(64)); double-buffered LDS GEMM
//   2) flash attention   -> f16 ctx + per-row (m,l); row-sum via ones-WMMA,
//                           row-max via v_permlane16, K/V tiles double-buffered
//   3) probability pass  -> attn = exp(S-m)/l straight to d_out (512MB, BW-bound)
//   4) output projection -> f32 out
//
// All LDS tiles padded to 72-half rows (144B): 16B-aligned (keeps ds_load_b128
// merging) and conflict-free fragment gathers (36n mod 64 injective for n<16).

typedef __attribute__((ext_vector_type(16))) _Float16 v16h;
typedef __attribute__((ext_vector_type(8)))  float    v8f;
typedef unsigned int u32;

#define B_   2
#define T1_  2048
#define T2_  2048
#define D_   1024
#define H_   16
#define HD_  64
#define LP   72       // padded LDS row stride in halfs
#define NEG_BIG (-3.0e38f)

#if defined(__has_builtin)
#  if __has_builtin(__builtin_amdgcn_permlane16)
#    define USE_PERMLANE16 1
#  endif
#endif
#ifndef USE_PERMLANE16
#  define USE_PERMLANE16 0
#endif

// ---------- WMMA fragment gathers from LDS (layouts per ISA 7.12.2) ----------
// A operand 16x32 f16 from row-major [rows][K] tile (strideH halfs/row).
// lane L: M=L&15, half=L>>4; VGPR i: K = 2i + 8*half (+8 if i>=4).
__device__ __forceinline__ v16h lds_loadA_16x32(const _Float16* t, int strideH,
                                                int rowBase, int colBase) {
  const int lane = threadIdx.x & 31;
  const int m = lane & 15, hh = lane >> 4;
  union { v16h h; u32 u[8]; } r;
  const u32* p = (const u32*)t;
  const int rowOff = (rowBase + m) * (strideH >> 1);
  #pragma unroll
  for (int i = 0; i < 8; ++i) {
    const int kk = 2 * i + 8 * hh + ((i >= 4) ? 8 : 0);
    r.u[i] = p[rowOff + ((colBase + kk) >> 1)];
  }
  return r.h;
}

// B operand 32x16 f16 (KxN) from tile stored TRANSPOSED as [N][K] (K contig).
// lane L: N=L&15, half=L>>4; VGPR j: K = 2j + 8*half (+8 if j>=4).
__device__ __forceinline__ v16h lds_loadB_32x16(const _Float16* t, int strideH,
                                                int nBase, int kBase) {
  const int lane = threadIdx.x & 31;
  const int n = lane & 15, hh = lane >> 4;
  union { v16h h; u32 u[8]; } r;
  const u32* p = (const u32*)t;
  const int rowOff = (nBase + n) * (strideH >> 1);
  #pragma unroll
  for (int j = 0; j < 8; ++j) {
    const int kk = 2 * j + 8 * hh + ((j >= 4) ? 8 : 0);
    r.u[j] = p[rowOff + ((kBase + kk) >> 1)];
  }
  return r.h;
}

__device__ __forceinline__ v8f wmma_f16(v16h a, v16h b, v8f c) {
  return __builtin_amdgcn_wmma_f32_16x16x32_f16(false, a, false, b,
                                                (short)0, c, false, false);
}

// ---------- row-max across the 16 lanes of a C-layout row (VALU only) -------
__device__ __forceinline__ float rowmax16(float x) {
#if USE_PERMLANE16
  // v_permlane16_b32 xor-tables; same table applies to lanes 16-31's row.
  #define PLX(s0, s1)                                                          \
    __uint_as_float(__builtin_amdgcn_permlane16(                               \
        __float_as_uint(x), __float_as_uint(x), (s0), (s1), false, false))
  x = fmaxf(x, PLX(0x67452301u, 0xEFCDAB89u));  // lane ^= 1
  x = fmaxf(x, PLX(0x54761032u, 0xDCFE98BAu));  // lane ^= 2
  x = fmaxf(x, PLX(0x32107654u, 0xBA98FEDCu));  // lane ^= 4
  x = fmaxf(x, PLX(0xFEDCBA98u, 0x76543210u));  // lane ^= 8
  #undef PLX
#else
  x = fmaxf(x, __shfl_xor(x, 1, 32));
  x = fmaxf(x, __shfl_xor(x, 2, 32));
  x = fmaxf(x, __shfl_xor(x, 4, 32));
  x = fmaxf(x, __shfl_xor(x, 8, 32));
#endif
  return x;
}

// ---------------- GEMM:  out[m,n] = (sum_k X[m,k]*W[k,n] + bias[n])*scale ----
// 256 threads (8 waves); block tile 128x64; K-step 64, double-buffered LDS.
template <typename XT, bool OUT_F16>
__global__ __launch_bounds__(256) void gemm_bias_kernel(
    const XT* __restrict__ X, const float* __restrict__ W,
    const float* __restrict__ bias, void* __restrict__ outv,
    int M, int K, int N, float scale) {
  __shared__ __align__(16) _Float16 Xs[2][128][LP];
  __shared__ __align__(16) _Float16 Ws[2][64][LP];  // [n][k]

  const int t = threadIdx.x, wave = t >> 5;
  const int lane = t & 31, nl = lane & 15, hh = lane >> 4;
  const int rowBlock = blockIdx.y * 128;
  const int colBlock = blockIdx.x * 64;

  auto stage = [&](int kb, int buf) {
    #pragma unroll
    for (int i = 0; i < 32; ++i) {  // 128x64 X tile, f32->f16
      int e = t + i * 256, r = e >> 6, c = e & 63;
      Xs[buf][r][c] = (_Float16)(X[(size_t)(rowBlock + r) * K + kb + c]);
    }
    #pragma unroll
    for (int i = 0; i < 16; ++i) {  // 64x64 W tile, stored [n][k]
      int e = t + i * 256, kr = e >> 6, nc = e & 63;
      Ws[buf][nc][kr] = (_Float16)(W[(size_t)(kb + kr) * N + colBlock + nc]);
    }
  };

  const v8f vzero = {0.f, 0.f, 0.f, 0.f, 0.f, 0.f, 0.f, 0.f};
  v8f acc[4] = {vzero, vzero, vzero, vzero};

  stage(0, 0);
  __syncthreads();
  const int NK = K / 64;
  for (int itk = 0; itk < NK; ++itk) {
    const int cur = itk & 1;
    if (itk + 1 < NK) stage((itk + 1) * 64, cur ^ 1);  // overlap next tile
    #pragma unroll
    for (int dc = 0; dc < 64; dc += 32) {
      v16h a = lds_loadA_16x32(&Xs[cur][0][0], LP, wave * 16, dc);
      #pragma unroll
      for (int j = 0; j < 4; ++j) {
        v16h b = lds_loadB_32x16(&Ws[cur][0][0], LP, j * 16, dc);
        acc[j] = wmma_f16(a, b, acc[j]);
      }
    }
    __syncthreads();
  }

  #pragma unroll
  for (int j = 0; j < 4; ++j) {
    const int col = colBlock + j * 16 + nl;
    const float bv = bias[col];
    #pragma unroll
    for (int r = 0; r < 8; ++r) {
      const int row = rowBlock + wave * 16 + r + 8 * hh;
      const float vv = (acc[j][r] + bv) * scale;
      if constexpr (OUT_F16)
        ((_Float16*)outv)[(size_t)row * N + col] = (_Float16)vv;
      else
        ((float*)outv)[(size_t)row * N + col] = vv;
    }
  }
}

// ---------------- Flash attention forward ------------------------------------
// grid (T1/128, B*H); 256 thr; k-tile 64, K/V double-buffered.
__global__ __launch_bounds__(256) void flash_fwd_kernel(
    const _Float16* __restrict__ q, const _Float16* __restrict__ k,
    const _Float16* __restrict__ v, const unsigned char* __restrict__ mask,
    _Float16* __restrict__ ctx, float* __restrict__ mstat,
    float* __restrict__ lstat) {
  __shared__ __align__(16) _Float16 Qs[128][LP];
  __shared__ __align__(16) _Float16 Ks[2][64][LP];
  __shared__ __align__(16) _Float16 Vt[2][64][LP];  // [hd][k]
  __shared__ __align__(16) _Float16 Ps[8][16][LP];
  __shared__ unsigned char Ms[T2_];

  const int t = threadIdx.x, wave = t >> 5, lane = t & 31;
  const int nl = lane & 15, hh = lane >> 4;
  const int bh = blockIdx.y, b = bh / H_, h = bh % H_;
  const int qRow0 = blockIdx.x * 128;
  const size_t qbase  = (size_t)(b * T1_ + qRow0) * D_ + h * HD_;
  const size_t kvbase = (size_t)(b * T2_) * D_ + h * HD_;

  #pragma unroll
  for (int i = 0; i < 16; ++i) {  // Q: 4096 dword pairs
    int e = t + i * 256, r = e >> 5, c2 = (e & 31) * 2;
    *(u32*)&Qs[r][c2] = *(const u32*)&q[qbase + (size_t)r * D_ + c2];
  }
  #pragma unroll
  for (int i = 0; i < 8; ++i)
    Ms[t + i * 256] = mask[(size_t)b * T2_ + t + i * 256];

  auto stageKV = [&](int it, int buf) {
    #pragma unroll
    for (int i = 0; i < 8; ++i) {  // 64x64 tiles as 2048 dword pairs
      int e = t + i * 256, r = e >> 5, c2 = (e & 31) * 2;
      const size_t gr = kvbase + (size_t)(it * 64 + r) * D_ + c2;
      *(u32*)&Ks[buf][r][c2] = *(const u32*)&k[gr];
      union { u32 u; _Float16 h[2]; } vv;
      vv.u = *(const u32*)&v[gr];
      Vt[buf][c2][r]     = vv.h[0];
      Vt[buf][c2 + 1][r] = vv.h[1];
    }
  };

  v16h ones;
  #pragma unroll
  for (int i = 0; i < 16; ++i) ones[i] = (_Float16)1.0f;

  float m_[8], l_[8];
  #pragma unroll
  for (int r = 0; r < 8; ++r) { m_[r] = NEG_BIG; l_[r] = 0.f; }
  const v8f vzero = {0.f, 0.f, 0.f, 0.f, 0.f, 0.f, 0.f, 0.f};
  v8f o[4] = {vzero, vzero, vzero, vzero};

  stageKV(0, 0);
  __syncthreads();

  const int NT = T2_ / 64;
  for (int it = 0; it < NT; ++it) {
    const int cur = it & 1;
    if (it + 1 < NT) stageKV(it + 1, cur ^ 1);  // overlap next K/V tile

    // S = Q K^T : 16x64 per wave (4 n-tiles), HD=64 in two K chunks
    v8f s[4] = {vzero, vzero, vzero, vzero};
    #pragma unroll
    for (int dc = 0; dc < HD_; dc += 32) {
      v16h a = lds_loadA_16x32(&Qs[0][0], LP, wave * 16, dc);
      #pragma unroll
      for (int j = 0; j < 4; ++j) {
        v16h bf = lds_loadB_32x16(&Ks[cur][0][0], LP, j * 16, dc);
        s[j] = wmma_f16(a, bf, s[j]);
      }
    }

    // key padding mask from LDS
    #pragma unroll
    for (int j = 0; j < 4; ++j) {
      const bool mk = Ms[it * 64 + j * 16 + nl] != 0;
      #pragma unroll
      for (int r = 0; r < 8; ++r)
        if (mk) s[j][r] = -1.0e9f;
    }

    // online softmax: row-max via permlane, row-sum via ones-WMMA below
    float sc_[8];
    #pragma unroll
    for (int r = 0; r < 8; ++r) {
      float vm = fmaxf(fmaxf(s[0][r], s[1][r]), fmaxf(s[2][r], s[3][r]));
      vm = rowmax16(vm);
      const float mn = fmaxf(m_[r], vm);
      sc_[r] = __expf(m_[r] - mn);
      m_[r] = mn;
    }

    // P = exp(S - m) -> per-wave LDS scratch (f16)
    #pragma unroll
    for (int r = 0; r < 8; ++r) {
      #pragma unroll
      for (int j = 0; j < 4; ++j)
        Ps[wave][r + 8 * hh][j * 16 + nl] = (_Float16)__expf(s[j][r] - m_[r]);
      o[0][r] *= sc_[r]; o[1][r] *= sc_[r]; o[2][r] *= sc_[r]; o[3][r] *= sc_[r];
    }

    // reload P as A fragments (same-wave LDS ops are in order)
    v16h pa0 = lds_loadA_16x32(&Ps[wave][0][0], LP, 0, 0);
    v16h pa1 = lds_loadA_16x32(&Ps[wave][0][0], LP, 0, 32);

    // row-sum = P x ones : lands exactly in C-layout rows, no shuffles
    v8f rs = wmma_f16(pa0, ones, vzero);
    rs = wmma_f16(pa1, ones, rs);

    // O += P V
    #pragma unroll
    for (int jt = 0; jt < 4; ++jt) {
      v16h b0 = lds_loadB_32x16(&Vt[cur][0][0], LP, jt * 16, 0);
      o[jt] = wmma_f16(pa0, b0, o[jt]);
      v16h b1 = lds_loadB_32x16(&Vt[cur][0][0], LP, jt * 16, 32);
      o[jt] = wmma_f16(pa1, b1, o[jt]);
    }
    #pragma unroll
    for (int r = 0; r < 8; ++r) l_[r] = l_[r] * sc_[r] + rs[r];

    __syncthreads();
  }

  #pragma unroll
  for (int r = 0; r < 8; ++r) {
    const int row = qRow0 + wave * 16 + r + 8 * hh;
    const float inv = 1.0f / l_[r];
    #pragma unroll
    for (int jt = 0; jt < 4; ++jt)
      ctx[(size_t)(b * T1_ + row) * D_ + h * HD_ + jt * 16 + nl] =
          (_Float16)(o[jt][r] * inv);
    if (nl == 0) {
      mstat[(size_t)bh * T1_ + row] = m_[r];
      lstat[(size_t)bh * T1_ + row] = l_[r];
    }
  }
}

// ---------------- Second pass: attn = exp(S - m)/l ---------------------------
__global__ __launch_bounds__(256) void attn_probs_kernel(
    const _Float16* __restrict__ q, const _Float16* __restrict__ k,
    const unsigned char* __restrict__ mask, const float* __restrict__ mstat,
    const float* __restrict__ lstat, float* __restrict__ attn) {
  __shared__ __align__(16) _Float16 Qs[128][LP];
  __shared__ __align__(16) _Float16 Ks[2][64][LP];
  __shared__ unsigned char Ms[T2_];

  const int t = threadIdx.x, wave = t >> 5, lane = t & 31;
  const int nl = lane & 15, hh = lane >> 4;
  const int bh = blockIdx.y, b = bh / H_, h = bh % H_;
  const int qRow0 = blockIdx.x * 128;
  const size_t qbase  = (size_t)(b * T1_ + qRow0) * D_ + h * HD_;
  const size_t kvbase = (size_t)(b * T2_) * D_ + h * HD_;

  #pragma unroll
  for (int i = 0; i < 16; ++i) {
    int e = t + i * 256, r = e >> 5, c2 = (e & 31) * 2;
    *(u32*)&Qs[r][c2] = *(const u32*)&q[qbase + (size_t)r * D_ + c2];
  }
  #pragma unroll
  for (int i = 0; i < 8; ++i)
    Ms[t + i * 256] = mask[(size_t)b * T2_ + t + i * 256];

  auto stageK = [&](int it, int buf) {
    #pragma unroll
    for (int i = 0; i < 8; ++i) {
      int e = t + i * 256, r = e >> 5, c2 = (e & 31) * 2;
      *(u32*)&Ks[buf][r][c2] =
          *(const u32*)&k[kvbase + (size_t)(it * 64 + r) * D_ + c2];
    }
  };

  float mrow[8], linv[8];
  #pragma unroll
  for (int r = 0; r < 8; ++r) {
    const int row = qRow0 + wave * 16 + r + 8 * hh;
    mrow[r] = mstat[(size_t)bh * T1_ + row];
    linv[r] = 1.0f / lstat[(size_t)bh * T1_ + row];
  }

  const v8f vzero = {0.f, 0.f, 0.f, 0.f, 0.f, 0.f, 0.f, 0.f};

  stageK(0, 0);
  __syncthreads();
  const int NT = T2_ / 64;
  for (int it = 0; it < NT; ++it) {
    const int cur = it & 1;
    if (it + 1 < NT) stageK(it + 1, cur ^ 1);

    v8f s[4] = {vzero, vzero, vzero, vzero};
    #pragma unroll
    for (int dc = 0; dc < HD_; dc += 32) {
      v16h a = lds_loadA_16x32(&Qs[0][0], LP, wave * 16, dc);
      #pragma unroll
      for (int j = 0; j < 4; ++j) {
        v16h bf = lds_loadB_32x16(&Ks[cur][0][0], LP, j * 16, dc);
        s[j] = wmma_f16(a, bf, s[j]);
      }
    }

    bool mk[4];
    #pragma unroll
    for (int j = 0; j < 4; ++j) mk[j] = Ms[it * 64 + j * 16 + nl] != 0;

    #pragma unroll
    for (int r = 0; r < 8; ++r) {
      const int row = qRow0 + wave * 16 + r + 8 * hh;
      const size_t base = ((size_t)bh * T1_ + row) * T2_ + it * 64;
      #pragma unroll
      for (int j = 0; j < 4; ++j) {
        const float sv = mk[j] ? -1.0e9f : s[j][r];
        attn[base + j * 16 + nl] = __expf(sv - mrow[r]) * linv[r];
      }
    }
    __syncthreads();
  }
}

// ---------------- host launcher ---------------------------------------------
extern "C" void kernel_launch(void* const* d_in, const int* in_sizes, int n_in,
                              void* d_out, int out_size, void* d_ws,
                              size_t ws_size, hipStream_t stream) {
  const float* query = (const float*)d_in[0];
  const float* key   = (const float*)d_in[1];
  const float* value = (const float*)d_in[2];
  const unsigned char* mask = (const unsigned char*)d_in[3];  // bool -> 1 byte
  const float* Wq_w = (const float*)d_in[4];
  const float* Wq_b = (const float*)d_in[5];
  const float* Wk_w = (const float*)d_in[6];
  const float* Wk_b = (const float*)d_in[7];
  const float* Wv_w = (const float*)d_in[8];
  const float* Wv_b = (const float*)d_in[9];
  const float* Wo_w = (const float*)d_in[10];
  const float* Wo_b = (const float*)d_in[11];

  char* ws = (char*)d_ws;
  const size_t f16buf = (size_t)B_ * T1_ * D_ * sizeof(_Float16);  // 8 MB
  _Float16* qh   = (_Float16*)ws;  ws += f16buf;
  _Float16* kh   = (_Float16*)ws;  ws += f16buf;
  _Float16* vh   = (_Float16*)ws;  ws += f16buf;
  _Float16* ctxh = (_Float16*)ws;  ws += f16buf;
  float* mstat = (float*)ws;       ws += (size_t)B_ * H_ * T1_ * 4;
  float* lstat = (float*)ws;

  float* out  = (float*)d_out;
  float* attn = out + (size_t)B_ * T1_ * D_;

  const dim3 blk(256);
  const dim3 gGemm(D_ / 64, (B_ * T1_) / 128);  // (16, 32)
  const dim3 gAttn(T1_ / 128, B_ * H_);         // (16, 32)

  gemm_bias_kernel<float, true><<<gGemm, blk, 0, stream>>>(
      query, Wq_w, Wq_b, qh, B_ * T1_, D_, D_, 0.125f);  // q * 1/sqrt(HD)
  gemm_bias_kernel<float, true><<<gGemm, blk, 0, stream>>>(
      key, Wk_w, Wk_b, kh, B_ * T2_, D_, D_, 1.0f);
  gemm_bias_kernel<float, true><<<gGemm, blk, 0, stream>>>(
      value, Wv_w, Wv_b, vh, B_ * T2_, D_, D_, 1.0f);

  flash_fwd_kernel<<<gAttn, blk, 0, stream>>>(qh, kh, vh, mask, ctxh, mstat,
                                              lstat);
  attn_probs_kernel<<<gAttn, blk, 0, stream>>>(qh, kh, mask, mstat, lstat,
                                               attn);

  gemm_bias_kernel<_Float16, false><<<gGemm, blk, 0, stream>>>(
      ctxh, Wo_w, Wo_b, out, B_ * T1_, D_, D_, 1.0f);
}
